// SpatialEncoder_15822659518637
// MI455X (gfx1250) — compile-verified, matrix operand order checked by reading, not run on hardware
//
#include <hip/hip_runtime.h>
#include <hip/hip_bf16.h>
#include <math.h>

#define GAT_HID 128
#define GAT_HEADS 8

typedef float    v2f  __attribute__((ext_vector_type(2)));
typedef float    v8f  __attribute__((ext_vector_type(8)));
typedef _Float16 v16h __attribute__((ext_vector_type(16)));

// ---------------------------------------------------------------------------
// GEMM: C[M x 128] = A[M x K] * B[K x 128] (+ bias), fp32 in/out via WMMA.
// One wave -> one 16x16 tile. 8 waves/block cover all 128 columns.
// ---------------------------------------------------------------------------
__global__ __launch_bounds__(256)
void gat_gemm_wmma(const float* __restrict__ A, const float* __restrict__ B,
                   const float* __restrict__ bias, float* __restrict__ C,
                   int M, int K) {
  const int lane = threadIdx.x & 31;
  const int wave = threadIdx.x >> 5;
  const int m0   = blockIdx.x * 16;
  const int n0   = wave * 16;          // 8 waves * 16 = 128 columns
  const int mr   = lane & 15;          // row within tile (and B column)
  const int kh   = lane >> 4;          // K-half selector (0/1)
  int arow = m0 + mr;  if (arow >= M) arow = M - 1;   // clamp loads, keep EXEC full
  v8f c = {};
#if __has_builtin(__builtin_amdgcn_wmma_f32_16x16x4_f32)
  for (int k0 = 0; k0 < K; k0 += 4) {
    const int ka = k0 + 2 * kh;
    v2f a, b;
    a[0] = A[(long long)arow * K + ka];
    a[1] = A[(long long)arow * K + ka + 1];
    b[0] = B[(long long)ka * GAT_HID + n0 + mr];
    b[1] = B[(long long)(ka + 1) * GAT_HID + n0 + mr];
    c = __builtin_amdgcn_wmma_f32_16x16x4_f32(false, a, false, b, (short)0, c,
                                              false, false);
  }
#else
  // Fallback: codegen-confirmed f16 WMMA (K step 32); fp32 converted on load.
  for (int k0 = 0; k0 < K; k0 += 32) {
    v16h a, b;
#pragma unroll
    for (int v = 0; v < 8; ++v) {
#pragma unroll
      for (int p = 0; p < 2; ++p) {
        const int kk = k0 + 8 * kh + ((v < 4) ? (2 * v + p) : (16 + 2 * (v - 4) + p));
        a[2 * v + p] = (_Float16)A[(long long)arow * K + kk];
        b[2 * v + p] = (_Float16)B[(long long)kk * GAT_HID + n0 + mr];
      }
    }
    c = __builtin_amdgcn_wmma_f32_16x16x32_f16(false, a, false, b, (short)0, c,
                                               false, false);
  }
#endif
  const float bv = bias ? bias[n0 + mr] : 0.0f;
#pragma unroll
  for (int r = 0; r < 8; ++r) {
    const int row = m0 + r + 8 * kh;   // C/D layout: VGPR r -> M = r (+8 for hi lanes)
    if (row < M) C[(long long)row * GAT_HID + n0 + mr] = c[r] + bv;
  }
}

// ---------------------------------------------------------------------------
// alpha_s/alpha_d: per (node, head) dot of h-segment with attention vectors.
// ---------------------------------------------------------------------------
__global__ void gat_alpha(const float* __restrict__ h,
                          const float* __restrict__ avs, const float* __restrict__ avd,
                          float* __restrict__ als, float* __restrict__ ald, int Nn) {
  const int t = blockIdx.x * blockDim.x + threadIdx.x;
  if (t >= Nn * GAT_HEADS) return;
  const int n = t >> 3, hd = t & 7;
  const float* ph = h + (long long)n * GAT_HID + hd * 16;
  const float* ps = avs + hd * 16;
  const float* pd = avd + hd * 16;
  float ss = 0.0f, sd = 0.0f;
#pragma unroll
  for (int j = 0; j < 16; ++j) {
    const float v = ph[j];
    ss += v * ps[j];
    sd += v * pd[j];
  }
  als[t] = ss;
  ald[t] = sd;
}

// monotonic int key for float atomic max (native global_atomic_max_i32)
__device__ __forceinline__ int f2key(float f) {
  const int i = __float_as_int(f);
  return i >= 0 ? i : (i ^ 0x7fffffff);
}
__device__ __forceinline__ float key2f(int k) {
  return __int_as_float(k >= 0 ? k : (k ^ 0x7fffffff));
}

__global__ void gat_init(float* __restrict__ outacc, float* __restrict__ denom,
                         int* __restrict__ emaxk, int Nn) {
  const long long t = (long long)blockIdx.x * blockDim.x + threadIdx.x;
  if (t < (long long)Nn * GAT_HID) outacc[t] = 0.0f;
  if (t < (long long)Nn * GAT_HEADS) {
    denom[t] = 0.0f;
    emaxk[t] = (int)0x807FFFFF;      // key(-inf)
  }
}

// pass 1: e = leakyrelu(a_s[src] + a_d[dst]); store e; segment max into emax[dst]
__global__ void gat_edge1(const long long* __restrict__ src, const long long* __restrict__ dst,
                          const float* __restrict__ als, const float* __restrict__ ald,
                          float* __restrict__ ebuf, int* __restrict__ emaxk, int E) {
  const int e = blockIdx.x * blockDim.x + threadIdx.x;
  if (e >= E) return;
  const long long s = src[e], d = dst[e];
  const float* ps = als + s * GAT_HEADS;
  const float* pd = ald + d * GAT_HEADS;
  float* pe = ebuf + (long long)e * GAT_HEADS;
  int* pm = emaxk + d * GAT_HEADS;
#pragma unroll
  for (int k = 0; k < GAT_HEADS; ++k) {
    float v = ps[k] + pd[k];
    v = v > 0.0f ? v : 0.2f * v;
    pe[k] = v;
    atomicMax(pm + k, f2key(v));
  }
}

// pass 2: ex = exp(e - emax[dst]); store ex; segment sum into denom[dst]
__global__ void gat_edge2(const long long* __restrict__ dst, float* __restrict__ ebuf,
                          const int* __restrict__ emaxk, float* __restrict__ denom, int E) {
  const int e = blockIdx.x * blockDim.x + threadIdx.x;
  if (e >= E) return;
  const long long d = dst[e];
  float* pe = ebuf + (long long)e * GAT_HEADS;
  const int* pm = emaxk + d * GAT_HEADS;
  float* pden = denom + d * GAT_HEADS;
#pragma unroll
  for (int k = 0; k < GAT_HEADS; ++k) {
    const float ex = __expf(pe[k] - key2f(pm[k]));
    pe[k] = ex;
    atomicAdd(pden + k, ex);
  }
}

// pass 3: one wave per edge; lane handles a float4 of the 128-wide message.
__global__ void gat_edge3(const long long* __restrict__ src, const long long* __restrict__ dst,
                          const float* __restrict__ ebuf, const float* __restrict__ denom,
                          const float* __restrict__ h, float* __restrict__ outacc, int E) {
  const long long gid = (long long)blockIdx.x * blockDim.x + threadIdx.x;
  const int e = (int)(gid >> 5);
  const int lane = (int)(gid & 31);
  if (e >= E) return;
  const long long s = src[e], d = dst[e];
  const int head = lane >> 2;                         // channel 4*lane -> head
  const float att = ebuf[(long long)e * GAT_HEADS + head] /
                    denom[d * GAT_HEADS + head];
  const float4 hv = *(const float4*)(h + s * GAT_HID + lane * 4);
  float* po = outacc + d * GAT_HID + lane * 4;
  atomicAdd(po + 0, att * hv.x);
  atomicAdd(po + 1, att * hv.y);
  atomicAdd(po + 2, att * hv.z);
  atomicAdd(po + 3, att * hv.w);
}

// epilogue: x = LayerNorm(elu(outacc + b_gat) + x) * g + b ; one wave per node
__global__ void gat_epilogue(const float* __restrict__ outacc, const float* __restrict__ bg,
                             const float* __restrict__ lng, const float* __restrict__ lnb,
                             float* __restrict__ x, int Nn) {
  const long long gid = (long long)blockIdx.x * blockDim.x + threadIdx.x;
  const int n = (int)(gid >> 5);
  const int lane = (int)(gid & 31);
  if (n >= Nn) return;
  const int c = lane * 4;
  const float4 o  = *(const float4*)(outacc + (long long)n * GAT_HID + c);
  const float4 bb = *(const float4*)(bg + c);
  const float4 r  = *(const float4*)(x + (long long)n * GAT_HID + c);
  float v[4] = {o.x + bb.x, o.y + bb.y, o.z + bb.z, o.w + bb.w};
  const float rr[4] = {r.x, r.y, r.z, r.w};
#pragma unroll
  for (int j = 0; j < 4; ++j) {
    v[j] = v[j] > 0.0f ? v[j] : (__expf(v[j]) - 1.0f);   // ELU
    v[j] += rr[j];                                       // residual
  }
  float s = v[0] + v[1] + v[2] + v[3];
#pragma unroll
  for (int off = 16; off > 0; off >>= 1) s += __shfl_xor(s, off, 32);
  const float mu = s * (1.0f / 128.0f);
  float q = 0.0f;
#pragma unroll
  for (int j = 0; j < 4; ++j) { v[j] -= mu; q += v[j] * v[j]; }
#pragma unroll
  for (int off = 16; off > 0; off >>= 1) q += __shfl_xor(q, off, 32);
  const float rs = rsqrtf(q * (1.0f / 128.0f) + 1e-5f);
  const float4 g  = *(const float4*)(lng + c);
  const float4 lb = *(const float4*)(lnb + c);
  float4 outv;
  outv.x = v[0] * rs * g.x + lb.x;
  outv.y = v[1] * rs * g.y + lb.y;
  outv.z = v[2] * rs * g.z + lb.z;
  outv.w = v[3] * rs * g.w + lb.w;
  *(float4*)(x + (long long)n * GAT_HID + c) = outv;
}

// ---------------------------------------------------------------------------
extern "C" void kernel_launch(void* const* d_in, const int* in_sizes, int n_in,
                              void* d_out, int out_size, void* d_ws, size_t ws_size,
                              hipStream_t stream) {
  const float*     nodef = (const float*)d_in[0];
  const long long* ei    = (const long long*)d_in[1];   // int64 edge index
  const float*     W_in  = (const float*)d_in[2];
  const float*     b_in  = (const float*)d_in[3];
  const float*     W     = (const float*)d_in[4];
  const float*     a_src = (const float*)d_in[5];
  const float*     a_dst = (const float*)d_in[6];
  const float*     b_gat = (const float*)d_in[7];
  const float*     ln_g  = (const float*)d_in[8];
  const float*     ln_b  = (const float*)d_in[9];

  const int Nn = in_sizes[0] / 64;     // F_IN = 64
  const int E  = in_sizes[1] / 2;      // edge_index is [2, E]
  const long long* src  = ei;
  const long long* dstv = ei + E;

  float* x = (float*)d_out;            // x lives in d_out throughout

  float* ws = (float*)d_ws;
  const long long NH = (long long)Nn * GAT_HID;
  float* h      = ws;
  float* outacc = ws + NH;
  float* als    = ws + 2 * NH;
  float* ald    = als + (long long)Nn * GAT_HEADS;
  float* denom  = ald + (long long)Nn * GAT_HEADS;
  float* ebuf   = denom + (long long)Nn * GAT_HEADS;
  int*   emaxk  = (int*)(ebuf + (long long)E * GAT_HEADS);

  const dim3 blk(256);
  const int mblocks = (Nn + 15) / 16;

  // input projection: x = nodef @ W_in + b_in
  gat_gemm_wmma<<<mblocks, blk, 0, stream>>>(nodef, W_in, b_in, x, Nn, 64);

  for (int L = 0; L < 2; ++L) {
    gat_gemm_wmma<<<mblocks, blk, 0, stream>>>(
        x, W + (long long)L * GAT_HID * GAT_HID, nullptr, h, Nn, GAT_HID);
    gat_alpha<<<(Nn * GAT_HEADS + 255) / 256, blk, 0, stream>>>(
        h, a_src + L * GAT_HID, a_dst + L * GAT_HID, als, ald, Nn);
    gat_init<<<(int)((NH + 255) / 256), blk, 0, stream>>>(outacc, denom, emaxk, Nn);
    gat_edge1<<<(E + 255) / 256, blk, 0, stream>>>(src, dstv, als, ald, ebuf, emaxk, E);
    gat_edge2<<<(E + 255) / 256, blk, 0, stream>>>(dstv, ebuf, emaxk, denom, E);
    gat_edge3<<<(int)(((long long)E * 32 + 255) / 256), blk, 0, stream>>>(
        src, dstv, ebuf, denom, h, outacc, E);
    gat_epilogue<<<(int)(((long long)Nn * 32 + 255) / 256), blk, 0, stream>>>(
        outacc, b_gat + L * GAT_HID, ln_g + L * GAT_HID, ln_b + L * GAT_HID, x, Nn);
  }
}
